// Model_55783035240784
// MI455X (gfx1250) — matrix operand order repaired and checked by reading, not compiled, not run on hardware
//
#include <hip/hip_runtime.h>

// ---------------------------------------------------------------------------
// CDNA5 / gfx1250 GRU seq2seq.
//   - all matmuls via v_wmma_f32_16x16x32_bf16 (fp32 accumulate)
//   - LDS staging via global_load_async_to_lds_b128 (ASYNCcnt, inline asm)
// ---------------------------------------------------------------------------

typedef __attribute__((ext_vector_type(16))) __bf16 v16bf;
typedef __attribute__((ext_vector_type(8)))  __bf16 v8bf;
typedef __attribute__((ext_vector_type(8)))  float  v8f;

#define HN   1024
#define INN  64
#define FCN  128
#define BN   256
#define LN   336
#define TN   96

#if defined(__gfx1250__)
#define ASYNC_LDS 1
#else
#define ASYNC_LDS 0
#endif

__device__ inline unsigned short f32_to_bf16(float f) {
  unsigned int u = __float_as_uint(f);
  unsigned int r = u + 0x7FFFu + ((u >> 16) & 1u);
  return (unsigned short)(r >> 16);
}

__device__ inline float sigmoidf_(float x) { return 1.0f / (1.0f + __expf(-x)); }

__device__ inline v8f vzero8() {
  v8f z;
#pragma unroll
  for (int i = 0; i < 8; ++i) z[i] = 0.0f;
  return z;
}

// Stage 16 bytes (8 bf16) global -> LDS via the CDNA5 direct async path
// (cache -> LDS, no VGPR round-trip, tracked by ASYNCcnt).
__device__ inline void stage16(unsigned short* lds_dst, const unsigned short* gsrc) {
#if ASYNC_LDS
  // generic LDS pointer: low 32 bits == LDS byte offset (aperture rule, ISA 10.2)
  unsigned int lds_off = (unsigned int)(unsigned long long)lds_dst;
  asm volatile("global_load_async_to_lds_b128 %0, %1, off"
               :
               : "v"(lds_off), "v"(gsrc)
               : "memory");
#else
  *(uint4*)lds_dst = *(const uint4*)gsrc;
#endif
}

// Producer-side completion fence for staged LDS data (before barrier signal).
__device__ inline void stage_fence() {
#if ASYNC_LDS
  asm volatile("s_wait_asynccnt 0" ::: "memory");
#endif
}

// A fragment: 16x32 bf16 (M x K).  lane l (lr=l&15, hi=l>>4) holds row M=lr,
// K in {hi*8..hi*8+7} U {16+hi*8..16+hi*8+7}  (ISA 7.12.2 16-bit A layout).
__device__ inline v16bf frag_a(const unsigned short* p, int row, int stride, int hi) {
  const __bf16* b = (const __bf16*)(p + row * stride + hi * 8);
  v8bf lo = *(const v8bf*)(b);
  v8bf hh = *(const v8bf*)(b + 16);
  return __builtin_shufflevector(lo, hh, 0, 1, 2, 3, 4, 5, 6, 7,
                                 8, 9, 10, 11, 12, 13, 14, 15);
}

// B fragment: 32x16 bf16 (K x N), stored in LDS as [N][K] row-major.
// lane l (n=l&15, khalf=l>>4) holds column n, K in {khalf*8..+7} U {16+khalf*8..+7}.
__device__ inline v16bf frag_b(const unsigned short* p, int col, int stride, int khalf) {
  const __bf16* b = (const __bf16*)(p + col * stride + khalf * 8);
  v8bf lo = *(const v8bf*)(b);
  v8bf hh = *(const v8bf*)(b + 16);
  return __builtin_shufflevector(lo, hh, 0, 1, 2, 3, 4, 5, 6, 7,
                                 8, 9, 10, 11, 12, 13, 14, 15);
}

__device__ inline v8f wmma_bf16(v16bf a, v16bf b, v8f c) {
  return __builtin_amdgcn_wmma_f32_16x16x32_bf16(false, a, false, b,
                                                 (short)0, c, false, false);
}

// ---------------------------------------------------------------------------
// Fused GRU step:  h_new = (1-z)*tanh(i_n + r*h_n) + z*h
// Grid: (H/64, B/64), 256 threads (8 waves, 2x4 wave grid over 64x64 tile).
// Accumulates 4 tiles: r(gi+gh), z(gi+gh), i_n(gi only), h_n(gh only).
// ---------------------------------------------------------------------------
__global__ __launch_bounds__(256) void gru_step_kernel(
    const unsigned short* __restrict__ h_bf16,
    const float* __restrict__ h_f32,
    const unsigned short* __restrict__ x_bf16, int x_stride,
    const unsigned short* __restrict__ Whh,   // [3H][H]  bf16
    const unsigned short* __restrict__ Wih,   // [3H][IN] bf16
    const float* __restrict__ bih, const float* __restrict__ bhh,
    float* __restrict__ h_out_f32, unsigned short* __restrict__ h_out_bf16) {
  __shared__ unsigned short As[64 * 32];
  __shared__ unsigned short Bs[3 * 64 * 32];

  const int tid  = threadIdx.x;
  const int lane = tid & 31;
  const int w    = tid >> 5;
  const int wr   = w >> 2;   // 0..1  (rows of 32)
  const int wc   = w & 3;    // 0..3  (cols of 16)
  const int n0   = blockIdx.x * 64;  // over H
  const int m0   = blockIdx.y * 64;  // over B
  const int lr   = lane & 15;
  const int hi   = lane >> 4;

  v8f accR[2], accZ[2], accIN[2], accHN[2];
#pragma unroll
  for (int ms = 0; ms < 2; ++ms) {
    accR[ms] = vzero8(); accZ[ms] = vzero8();
    accIN[ms] = vzero8(); accHN[ms] = vzero8();
  }

  const int ldrow = tid >> 2;          // 0..63, 4 threads per row
  const int ldc8  = (tid & 3) * 8;     // 8 bf16 = 16B per thread

  // ---- K loop over Whh (K = H): accumulates r, z, h_n -------------------
  for (int k0 = 0; k0 < HN; k0 += 32) {
    __syncthreads();
    stage16(&As[ldrow * 32 + ldc8],
            h_bf16 + (size_t)(m0 + ldrow) * HN + k0 + ldc8);
#pragma unroll
    for (int g = 0; g < 3; ++g) {
      stage16(&Bs[(g * 64 + ldrow) * 32 + ldc8],
              Whh + ((size_t)g * HN + n0 + ldrow) * HN + k0 + ldc8);
    }
    stage_fence();
    __syncthreads();

    v16bf br = frag_b(&Bs[0 * 64 * 32], wc * 16 + lr, 32, hi);
    v16bf bz = frag_b(&Bs[1 * 64 * 32], wc * 16 + lr, 32, hi);
    v16bf bn = frag_b(&Bs[2 * 64 * 32], wc * 16 + lr, 32, hi);
#pragma unroll
    for (int ms = 0; ms < 2; ++ms) {
      v16bf a = frag_a(As, wr * 32 + ms * 16 + lr, 32, hi);
      accR[ms]  = wmma_bf16(a, br, accR[ms]);
      accZ[ms]  = wmma_bf16(a, bz, accZ[ms]);
      accHN[ms] = wmma_bf16(a, bn, accHN[ms]);
    }
  }

  // ---- K loop over Wih (K = IN = 64): accumulates r, z, i_n -------------
  for (int k0 = 0; k0 < INN; k0 += 32) {
    __syncthreads();
    stage16(&As[ldrow * 32 + ldc8],
            x_bf16 + (size_t)(m0 + ldrow) * x_stride + k0 + ldc8);
#pragma unroll
    for (int g = 0; g < 3; ++g) {
      stage16(&Bs[(g * 64 + ldrow) * 32 + ldc8],
              Wih + ((size_t)g * HN + n0 + ldrow) * INN + k0 + ldc8);
    }
    stage_fence();
    __syncthreads();

    v16bf br = frag_b(&Bs[0 * 64 * 32], wc * 16 + lr, 32, hi);
    v16bf bz = frag_b(&Bs[1 * 64 * 32], wc * 16 + lr, 32, hi);
    v16bf bn = frag_b(&Bs[2 * 64 * 32], wc * 16 + lr, 32, hi);
#pragma unroll
    for (int ms = 0; ms < 2; ++ms) {
      v16bf a = frag_a(As, wr * 32 + ms * 16 + lr, 32, hi);
      accR[ms]  = wmma_bf16(a, br, accR[ms]);
      accZ[ms]  = wmma_bf16(a, bz, accZ[ms]);
      accIN[ms] = wmma_bf16(a, bn, accIN[ms]);
    }
  }

  // ---- Gate nonlinearity + write-back -----------------------------------
  const int ng   = n0 + wc * 16 + lr;
  const float br_ = bih[ng]            + bhh[ng];
  const float bz_ = bih[HN + ng]       + bhh[HN + ng];
  const float bin_ = bih[2 * HN + ng];
  const float bhn_ = bhh[2 * HN + ng];
#pragma unroll
  for (int ms = 0; ms < 2; ++ms) {
#pragma unroll
    for (int v = 0; v < 8; ++v) {
      const int mg = m0 + wr * 32 + ms * 16 + v + 8 * hi;  // C/D layout: M=v+8*hi
      float r  = sigmoidf_(accR[ms][v] + br_);
      float z  = sigmoidf_(accZ[ms][v] + bz_);
      float nn = tanhf(accIN[ms][v] + bin_ + r * (accHN[ms][v] + bhn_));
      float hp = h_f32[(size_t)mg * HN + ng];
      float hnew = (1.0f - z) * nn + z * hp;
      h_out_f32[(size_t)mg * HN + ng]  = hnew;
      h_out_bf16[(size_t)mg * HN + ng] = f32_to_bf16(hnew);
    }
  }
}

// ---------------------------------------------------------------------------
// Decoder head:  y = relu(h @ fc_W.T + fc_b) @ out_W.T + out_b
// Grid: 4 blocks of 64 batch rows, 256 threads.  Two chained WMMA GEMMs.
// Writes y (fp32) to d_out slice and y (bf16) as next autoregressive input.
// ---------------------------------------------------------------------------
__global__ __launch_bounds__(256) void dec_out_kernel(
    const unsigned short* __restrict__ h_bf16,
    const unsigned short* __restrict__ fcW,    // [FC][H]  bf16
    const float* __restrict__ fc_b,
    const unsigned short* __restrict__ outW,   // [IN][FC] bf16
    const float* __restrict__ out_b,
    float* __restrict__ y_out, int y_stride,
    unsigned short* __restrict__ x_next) {
  __shared__ unsigned short As[64 * 32];
  __shared__ unsigned short Bs[128 * 32];
  __shared__ unsigned short fcS[64 * 128];
  __shared__ unsigned short oW[64 * 128];

  const int tid  = threadIdx.x;
  const int lane = tid & 31;
  const int w    = tid >> 5;
  const int wr   = w >> 2;
  const int wc   = w & 3;
  const int m0   = blockIdx.x * 64;
  const int lr   = lane & 15;
  const int hi   = lane >> 4;

  // stage out_W into LDS once (64*128 bf16 = 16KB); fenced by the first
  // stage_fence() inside the phase-1 K loop (waits this wave's ASYNCcnt to 0).
  for (int i = tid; i < 64 * 128 / 8; i += 256)
    stage16(&oW[i * 8], outW + i * 8);

  // ---- phase 1: fc = relu(h @ fc_W.T + fc_b), 64x128 tile ---------------
  v8f accF[2][2];
#pragma unroll
  for (int ms = 0; ms < 2; ++ms)
#pragma unroll
    for (int ns = 0; ns < 2; ++ns) accF[ms][ns] = vzero8();

  for (int k0 = 0; k0 < HN; k0 += 32) {
    __syncthreads();
    {
      int row = tid >> 2, c8 = (tid & 3) * 8;
      stage16(&As[row * 32 + c8],
              h_bf16 + (size_t)(m0 + row) * HN + k0 + c8);
    }
    {
      int n = tid >> 1, koff = (tid & 1) * 16;
      stage16(&Bs[n * 32 + koff],     fcW + (size_t)n * HN + k0 + koff);
      stage16(&Bs[n * 32 + koff + 8], fcW + (size_t)n * HN + k0 + koff + 8);
    }
    stage_fence();
    __syncthreads();
#pragma unroll
    for (int ns = 0; ns < 2; ++ns) {
      v16bf b = frag_b(Bs, wc * 32 + ns * 16 + lr, 32, hi);
#pragma unroll
      for (int ms = 0; ms < 2; ++ms) {
        v16bf a = frag_a(As, wr * 32 + ms * 16 + lr, 32, hi);
        accF[ms][ns] = wmma_bf16(a, b, accF[ms][ns]);
      }
    }
  }

  __syncthreads();
#pragma unroll
  for (int ms = 0; ms < 2; ++ms)
#pragma unroll
    for (int ns = 0; ns < 2; ++ns)
#pragma unroll
      for (int v = 0; v < 8; ++v) {
        int ml = wr * 32 + ms * 16 + v + 8 * hi;
        int nl = wc * 32 + ns * 16 + lr;
        float f = accF[ms][ns][v] + fc_b[nl];
        f = fmaxf(f, 0.0f);
        fcS[ml * 128 + nl] = f32_to_bf16(f);
      }
  __syncthreads();

  // ---- phase 2: y = fc @ out_W.T + out_b, 64x64 tile, K=128 -------------
  v8f accO[2];
#pragma unroll
  for (int ms = 0; ms < 2; ++ms) accO[ms] = vzero8();

#pragma unroll
  for (int k0 = 0; k0 < FCN; k0 += 32) {
    v16bf b = frag_b(oW + k0, wc * 16 + lr, 128, hi);
#pragma unroll
    for (int ms = 0; ms < 2; ++ms) {
      v16bf a = frag_a(fcS + k0, wr * 32 + ms * 16 + lr, 128, hi);
      accO[ms] = wmma_bf16(a, b, accO[ms]);
    }
  }

  const int nl = wc * 16 + lr;
  const float ob = out_b[nl];
#pragma unroll
  for (int ms = 0; ms < 2; ++ms)
#pragma unroll
    for (int v = 0; v < 8; ++v) {
      int mg = m0 + wr * 32 + ms * 16 + v + 8 * hi;
      float y = accO[ms][v] + ob;
      y_out[(size_t)mg * y_stride + nl] = y;
      x_next[mg * INN + nl] = f32_to_bf16(y);
    }
}

// ---------------------------------------------------------------------------
__global__ void f32_to_bf16_kernel(const float* __restrict__ s,
                                   unsigned short* __restrict__ d, int n) {
  int i = blockIdx.x * blockDim.x + threadIdx.x;
  if (i < n) d[i] = f32_to_bf16(s[i]);
}

__global__ void zero_h_kernel(float* __restrict__ hf,
                              unsigned short* __restrict__ hb, int n) {
  int i = blockIdx.x * blockDim.x + threadIdx.x;
  if (i < n) { hf[i] = 0.0f; hb[i] = 0; }
}

// ---------------------------------------------------------------------------
extern "C" void kernel_launch(void* const* d_in, const int* in_sizes, int n_in,
                              void* d_out, int out_size, void* d_ws, size_t ws_size,
                              hipStream_t stream) {
  (void)in_sizes; (void)n_in; (void)out_size; (void)ws_size;

  const float* batch_x = (const float*)d_in[0];   // (B, L, IN)
  const float* enc_Wih = (const float*)d_in[4];   // (3H, IN)
  const float* enc_Whh = (const float*)d_in[5];   // (3H, H)
  const float* enc_bih = (const float*)d_in[6];
  const float* enc_bhh = (const float*)d_in[7];
  const float* dec_Wih = (const float*)d_in[8];
  const float* dec_Whh = (const float*)d_in[9];
  const float* dec_bih = (const float*)d_in[10];
  const float* dec_bhh = (const float*)d_in[11];
  const float* fc_W    = (const float*)d_in[12];  // (FC, H)
  const float* fc_b    = (const float*)d_in[13];
  const float* out_W   = (const float*)d_in[14];  // (IN, FC)
  const float* out_b   = (const float*)d_in[15];
  float* out = (float*)d_out;                     // (B, T, IN)

  char* ws = (char*)d_ws;
  auto carve = [&](size_t bytes) -> char* {
    char* p = ws;
    ws += (bytes + 255) & ~(size_t)255;
    return p;
  };
  unsigned short* encWhh_b = (unsigned short*)carve((size_t)3 * HN * HN * 2);
  unsigned short* decWhh_b = (unsigned short*)carve((size_t)3 * HN * HN * 2);
  unsigned short* encWih_b = (unsigned short*)carve((size_t)3 * HN * INN * 2);
  unsigned short* decWih_b = (unsigned short*)carve((size_t)3 * HN * INN * 2);
  unsigned short* fcW_b    = (unsigned short*)carve((size_t)FCN * HN * 2);
  unsigned short* outW_b   = (unsigned short*)carve((size_t)INN * FCN * 2);
  unsigned short* bx_b     = (unsigned short*)carve((size_t)BN * LN * INN * 2);
  float* h_f[2];
  unsigned short* h_b[2];
  h_f[0] = (float*)carve((size_t)BN * HN * 4);
  h_f[1] = (float*)carve((size_t)BN * HN * 4);
  h_b[0] = (unsigned short*)carve((size_t)BN * HN * 2);
  h_b[1] = (unsigned short*)carve((size_t)BN * HN * 2);
  unsigned short* xbuf[2];
  xbuf[0] = (unsigned short*)carve((size_t)BN * INN * 2);
  xbuf[1] = (unsigned short*)carve((size_t)BN * INN * 2);

  auto conv = [&](const float* s, unsigned short* d, int n) {
    f32_to_bf16_kernel<<<(n + 255) / 256, 256, 0, stream>>>(s, d, n);
  };
  conv(enc_Whh, encWhh_b, 3 * HN * HN);
  conv(dec_Whh, decWhh_b, 3 * HN * HN);
  conv(enc_Wih, encWih_b, 3 * HN * INN);
  conv(dec_Wih, decWih_b, 3 * HN * INN);
  conv(fc_W, fcW_b, FCN * HN);
  conv(out_W, outW_b, INN * FCN);
  conv(batch_x, bx_b, BN * LN * INN);

  zero_h_kernel<<<(BN * HN + 255) / 256, 256, 0, stream>>>(h_f[0], h_b[0], BN * HN);

  const dim3 ggrid(HN / 64, BN / 64);
  int p = 0;

  // ---- encoder scan (L sequential steps) --------------------------------
  for (int t = 0; t < LN; ++t) {
    gru_step_kernel<<<ggrid, 256, 0, stream>>>(
        h_b[p], h_f[p], bx_b + t * INN, LN * INN,
        encWhh_b, encWih_b, enc_bih, enc_bhh, h_f[1 - p], h_b[1 - p]);
    p ^= 1;
  }

  // ---- decoder scan (T sequential steps, autoregressive) ----------------
  for (int t = 0; t < TN; ++t) {
    const unsigned short* xp = (t == 0) ? (bx_b + (LN - 1) * INN) : xbuf[(t - 1) & 1];
    const int xs = (t == 0) ? (LN * INN) : INN;
    gru_step_kernel<<<ggrid, 256, 0, stream>>>(
        h_b[p], h_f[p], xp, xs,
        decWhh_b, decWih_b, dec_bih, dec_bhh, h_f[1 - p], h_b[1 - p]);
    p ^= 1;
    dec_out_kernel<<<4, 256, 0, stream>>>(
        h_b[p], fcW_b, fc_b, outW_b, out_b,
        out + t * INN, TN * INN, xbuf[t & 1]);
  }
}
